// SwinTransformerLayer_34351148433517
// MI455X (gfx1250) — matrix operand order repaired and checked by reading, not compiled, not run on hardware
//
#include <hip/hip_runtime.h>
#include <hip/hip_bf16.h>
#include <math.h>

// ---------------------------------------------------------------------------
// Swin layer (with replicated source bugs) for gfx1250 / MI455X, wave32 WMMA.
//   x1 = winrev(MHA(winpart(LN1(x)))) + x
//   x2 = MLP1(LN2(x1)) + x            (residual bug: original x)
//   x3 = P(LN3(x2)) + x2              (SW-MSA output discarded -> pure gather)
//   out= MLP2(LN4(x3)) + x3
// Data movement: activation tiles staged into LDS with
// GLOBAL_LOAD_ASYNC_TO_LDS_B128 (ASYNCcnt), GEMMs in f16 WMMA, f32 accum.
// ---------------------------------------------------------------------------

typedef __attribute__((ext_vector_type(16))) _Float16 v16h;
typedef __attribute__((ext_vector_type(8)))  float    v8f;

#define LN_EPS 1e-5f
#define NTOK   (64 * 3136)          // B * L
#define XBYTES ((size_t)NTOK * 96 * sizeof(float))

__device__ __forceinline__ v8f wmma_f16(v16h a, v16h b, v8f c) {
  return __builtin_amdgcn_wmma_f32_16x16x32_f16(false, a, false, b, (short)0, c,
                                                false, false);
}

// Per-lane async copy of 16 bytes global -> LDS (ASYNCcnt tracked, no VGPR data)
__device__ __forceinline__ void async_g2l_b128(uint32_t lds_off,
                                               const void* gptr) {
  uint64_t ga = (uint64_t)(uintptr_t)gptr;
  asm volatile("global_load_async_to_lds_b128 %0, %1, off"
               :: "v"(lds_off), "v"(ga) : "memory");
}
__device__ __forceinline__ void wait_async0() {
  asm volatile("s_wait_asynccnt 0" ::: "memory");
}
// Low 32 bits of a flat LDS pointer are the LDS byte address (ISA 10.2).
__device__ __forceinline__ uint32_t lds_addr32(const void* p) {
  return (uint32_t)(uintptr_t)p;
}

// A fragment (16x32 tile of row-major M x K, ld in halves).
// Layout: lane 0-15 row m, elems 0-15 -> K=16*hi+e (matches B fragments).
__device__ __forceinline__ v16h load_A(const _Float16* src, int ld, int mtile,
                                       int kbase, int lane) {
  int m  = mtile * 16 + (lane & 15);
  int hi = lane >> 4;
  const _Float16* row = src + (size_t)m * ld + kbase;
  v16h a;
#pragma unroll
  for (int e = 0; e < 16; ++e) a[e] = row[16 * hi + e];
  return a;
}

// B fragment where B[k][n] = W[n][k] (W row-major N x K) -> computes X @ W^T.
// Contiguous k order (same as A): lane group hi holds K = kbase + 16*hi + e, contiguous in W row n.
__device__ __forceinline__ v16h load_B_T(const _Float16* W, int ld, int ntile,
                                         int kbase, int lane) {
  int n  = ntile * 16 + (lane & 15);
  int hi = lane >> 4;
  const _Float16* row = W + (size_t)n * ld + kbase + 16 * hi;
  v16h b;
#pragma unroll
  for (int e = 0; e < 16; ++e) b[e] = row[e];
  return b;
}

// B fragment where B[k][n] = V[k][n] (V row-major K x N).
__device__ __forceinline__ v16h load_B_N(const _Float16* V, int ld, int ntile,
                                         int kbase, int lane) {
  int n  = ntile * 16 + (lane & 15);
  int hi = lane >> 4;
  v16h b;
#pragma unroll
  for (int e = 0; e < 16; ++e) b[e] = V[(size_t)(kbase + 16 * hi + e) * ld + n];
  return b;
}

// ---------------------------------------------------------------------------
// Weight f32 -> f16 conversion.
// ---------------------------------------------------------------------------
__global__ void cvt_f16_kernel(const float* __restrict__ in,
                               _Float16* __restrict__ out, int n) {
  int i = blockIdx.x * blockDim.x + threadIdx.x;
  if (i < n) out[i] = (_Float16)in[i];
}

// ---------------------------------------------------------------------------
// Kernel 1: fused LN1 + window attention + proj + residual. One block per
// 7x7 window (49 tokens padded to 64 rows). 128 threads = 4 waves; wave w
// owns M-tile w (rows 16w..16w+15) of every GEMM.
// LDS: [ sm_x 12288 | sm_qkv 36864 (staging aliased here) | sm_p 8192 ]
// ---------------------------------------------------------------------------
__global__ __launch_bounds__(128) void win_attn_kernel(
    const float* __restrict__ x, const float* __restrict__ ln_g,
    const float* __restrict__ ln_b, const _Float16* __restrict__ qkv_wh,
    const float* __restrict__ qkv_b, const _Float16* __restrict__ proj_wh,
    const float* __restrict__ proj_b, float* __restrict__ x1) {
  __shared__ __align__(16) char smem[64 * 96 * 2 + 64 * 288 * 2 + 64 * 64 * 2];
  _Float16* sm_x   = (_Float16*)smem;                  // LN'd input, later O
  _Float16* sm_qkv = (_Float16*)(smem + 64 * 96 * 2);  // Q|K|V
  _Float16* sm_p   = (_Float16*)(smem + 64 * 96 * 2 + 64 * 288 * 2); // probs
  float*    stage  = (float*)sm_qkv;  // 49x96 f32 staging, dead before QKV use

  const int widx = blockIdx.x;          // 0..4095
  const int b    = widx >> 6;
  const int wi   = widx & 63;
  const int wr   = wi >> 3, wc = wi & 7;
  const size_t tb = (size_t)b * 3136;   // token base of this batch image

  const int tid  = threadIdx.x;
  const int wv   = tid >> 5;            // wave id = M-tile
  const int lane = tid & 31;
  const int nidx = lane & 15;
  const int hi   = lane >> 4;

  // ---- Phase -1: async-stage the 49 window rows (96 f32 each) into LDS ----
  {
    uint32_t lbase = lds_addr32(stage);
    for (int c = tid; c < 49 * 24; c += 128) {   // 24 x 16B chunks per row
      int row = c / 24, ch = c - row * 24;
      int i = row / 7, j = row - 7 * i;
      const float* g =
          x + (tb + (size_t)(wr * 7 + i) * 56 + wc * 7 + j) * 96 + ch * 4;
      async_g2l_b128(lbase + (uint32_t)(row * 96 + ch * 4) * 4, g);
    }
    wait_async0();
  }
  __syncthreads();

  // ---- Phase 0: LayerNorm rows into LDS (f16), zero the 15 pad rows ----
  if (tid < 64) {
    if (tid < 49) {
      const float* xr = stage + tid * 96;
      float s = 0.f, ss = 0.f;
      for (int k = 0; k < 96; ++k) { float v = xr[k]; s += v; ss += v * v; }
      float mean = s * (1.f / 96.f);
      float inv  = rsqrtf(ss * (1.f / 96.f) - mean * mean + LN_EPS);
      for (int k = 0; k < 96; ++k)
        sm_x[tid * 96 + k] = (_Float16)((xr[k] - mean) * inv * ln_g[k] + ln_b[k]);
    } else {
      for (int k = 0; k < 96; ++k) sm_x[tid * 96 + k] = (_Float16)0.f;
    }
  }
  __syncthreads();

  // ---- Phase 1: QKV = Xln @ Wqkv^T + b  (64x96 @ 96x288) ----
  for (int nt = 0; nt < 18; ++nt) {
    v8f acc = {};
#pragma unroll
    for (int kc = 0; kc < 3; ++kc) {
      v16h a  = load_A(sm_x, 96, wv, 32 * kc, lane);
      v16h bf = load_B_T(qkv_wh, 96, nt, 32 * kc, lane);
      acc = wmma_f16(a, bf, acc);
    }
    int   n    = nt * 16 + nidx;
    float bias = qkv_b[n];
#pragma unroll
    for (int e = 0; e < 8; ++e)
      sm_qkv[(wv * 16 + hi * 8 + e) * 288 + n] = (_Float16)(acc[e] + bias);
  }
  __syncthreads();

  // ---- Phase 2: per-head attention ----
  const float scale = 0.17677669529663687f; // 1/sqrt(32)
  for (int h = 0; h < 3; ++h) {
    // scores S = Q @ K^T: wave's 16 rows x 64 cols, K-dim = 32 (one chunk)
    v8f sc[4];
#pragma unroll
    for (int nt = 0; nt < 4; ++nt) {
      v8f acc = {};
      v16h a  = load_A(sm_qkv + h * 32, 288, wv, 0, lane);
      v16h bf = load_B_T(sm_qkv + 96 + h * 32, 288, nt, 0, lane);
      sc[nt] = wmma_f16(a, bf, acc);
    }
    // in-register row softmax (cols >= 49 masked); cross-lane reduce within
    // each 16-lane half (xor masks 8..1 never cross the 16-lane boundary)
#pragma unroll
    for (int e = 0; e < 8; ++e) {
      float v0[4];
      float mx = -INFINITY;
#pragma unroll
      for (int nt = 0; nt < 4; ++nt) {
        int   col = nt * 16 + nidx;
        float v   = sc[nt][e] * scale;
        v0[nt]    = (col < 49) ? v : -INFINITY;
        mx        = fmaxf(mx, v0[nt]);
      }
      for (int m = 8; m >= 1; m >>= 1) mx = fmaxf(mx, __shfl_xor(mx, m, 32));
      float p[4], sum = 0.f;
#pragma unroll
      for (int nt = 0; nt < 4; ++nt) { p[nt] = expf(v0[nt] - mx); sum += p[nt]; }
      for (int m = 8; m >= 1; m >>= 1) sum += __shfl_xor(sum, m, 32);
      float rinv = 1.f / sum;
      int   row  = wv * 16 + hi * 8 + e;
#pragma unroll
      for (int nt = 0; nt < 4; ++nt)
        sm_p[row * 64 + nt * 16 + nidx] = (_Float16)(p[nt] * rinv);
    }
    __syncthreads();

    // O_h = P @ V : 64x64 @ 64x32 (K-dim 64 -> two chunks)
#pragma unroll
    for (int nt = 0; nt < 2; ++nt) {
      v8f acc = {};
#pragma unroll
      for (int kc = 0; kc < 2; ++kc) {
        v16h a  = load_A(sm_p, 64, wv, 32 * kc, lane);
        v16h bf = load_B_N(sm_qkv + 192 + h * 32, 288, nt, 32 * kc, lane);
        acc = wmma_f16(a, bf, acc);
      }
      int n = h * 32 + nt * 16 + nidx;
#pragma unroll
      for (int e = 0; e < 8; ++e)
        sm_x[(wv * 16 + hi * 8 + e) * 96 + n] = (_Float16)acc[e];
    }
    __syncthreads();
  }

  // ---- Phase 3: proj + bias + residual, window_reverse store ----
  for (int nt = 0; nt < 6; ++nt) {
    v8f acc = {};
#pragma unroll
    for (int kc = 0; kc < 3; ++kc) {
      v16h a  = load_A(sm_x, 96, wv, 32 * kc, lane);
      v16h bf = load_B_T(proj_wh, 96, nt, 32 * kc, lane);
      acc = wmma_f16(a, bf, acc);
    }
    int   n    = nt * 16 + nidx;
    float bias = proj_b[n];
#pragma unroll
    for (int e = 0; e < 8; ++e) {
      int m = wv * 16 + hi * 8 + e;
      if (m < 49) {
        int i = m / 7, j = m - 7 * i;
        size_t gi = (tb + (size_t)(wr * 7 + i) * 56 + wc * 7 + j) * 96 + n;
        x1[gi] = acc[e] + bias + x[gi];
      }
    }
  }
}

// ---------------------------------------------------------------------------
// Kernel 2/4: fused LN + MLP (96->384 GELU(erf) ->96) + residual.
// One block per 64-token tile; 4 waves, wave w owns M-tile w.
// LDS: [ sx 12288 | sh 49152 (f32 staging aliased onto sh) ]
// ---------------------------------------------------------------------------
__global__ __launch_bounds__(128) void mlp_kernel(
    const float* __restrict__ xin, const float* __restrict__ res,
    const float* __restrict__ ln_g, const float* __restrict__ ln_b,
    const _Float16* __restrict__ w1h, const float* __restrict__ b1,
    const _Float16* __restrict__ w2h, const float* __restrict__ b2,
    float* __restrict__ out) {
  __shared__ __align__(16) char smem[64 * 96 * 2 + 64 * 384 * 2];
  _Float16* sx = (_Float16*)smem;
  _Float16* sh = (_Float16*)(smem + 64 * 96 * 2);
  float* stage = (float*)sh;            // 64x96 f32 staging, dead before gemm1

  const size_t base = (size_t)blockIdx.x * 64;
  const int tid  = threadIdx.x;
  const int wv   = tid >> 5;
  const int lane = tid & 31;
  const int nidx = lane & 15;
  const int hi   = lane >> 4;

  __builtin_prefetch(w1h, 0, 0);
  __builtin_prefetch(w2h, 0, 0);

  // ---- async-stage the 64x96 f32 activation tile into LDS ----
  {
    uint32_t lbase = lds_addr32(stage);
    const float* g0 = xin + base * 96;
#pragma unroll
    for (int it = 0; it < 12; ++it) {    // 1536 x 16B chunks / 128 threads
      int c = it * 128 + tid;
      async_g2l_b128(lbase + (uint32_t)c * 16, g0 + c * 4);
    }
    wait_async0();
  }
  __syncthreads();

  if (tid < 64) {
    const float* xr = stage + tid * 96;
    float s = 0.f, ss = 0.f;
    for (int k = 0; k < 96; ++k) { float v = xr[k]; s += v; ss += v * v; }
    float mean = s * (1.f / 96.f);
    float inv  = rsqrtf(ss * (1.f / 96.f) - mean * mean + LN_EPS);
    for (int k = 0; k < 96; ++k)
      sx[tid * 96 + k] = (_Float16)((xr[k] - mean) * inv * ln_g[k] + ln_b[k]);
  }
  __syncthreads();

  // H = GELU(Xln @ W1^T + b1) : 64x96 @ 96x384
  for (int nt = 0; nt < 24; ++nt) {
    v8f acc = {};
#pragma unroll
    for (int kc = 0; kc < 3; ++kc) {
      v16h a  = load_A(sx, 96, wv, 32 * kc, lane);
      v16h bf = load_B_T(w1h, 96, nt, 32 * kc, lane);
      acc = wmma_f16(a, bf, acc);
    }
    int   n  = nt * 16 + nidx;
    float bi = b1[n];
#pragma unroll
    for (int e = 0; e < 8; ++e) {
      float v = acc[e] + bi;
      float g = 0.5f * v * (1.f + erff(v * 0.70710678118654752f));
      sh[(wv * 16 + hi * 8 + e) * 384 + n] = (_Float16)g;
    }
  }
  __syncthreads();

  // out = H @ W2^T + b2 + res : 64x384 @ 384x96
  for (int nt = 0; nt < 6; ++nt) {
    v8f acc = {};
    for (int kc = 0; kc < 12; ++kc) {
      v16h a  = load_A(sh, 384, wv, 32 * kc, lane);
      v16h bf = load_B_T(w2h, 384, nt, 32 * kc, lane);
      acc = wmma_f16(a, bf, acc);
    }
    int   n  = nt * 16 + nidx;
    float bi = b2[n];
#pragma unroll
    for (int e = 0; e < 8; ++e) {
      size_t gi = (base + wv * 16 + hi * 8 + e) * 96 + n;
      out[gi] = acc[e] + bi + res[gi];
    }
  }
}

// ---------------------------------------------------------------------------
// Kernel 3: x3[t] = LN3(x2[perm(t)]) + x2[t]   (SW-MSA branch collapses to a
// pure gather: window_reverse(raw reshape) followed by roll(+3,+3))
// ---------------------------------------------------------------------------
__global__ void permute_ln_kernel(const float* __restrict__ x2,
                                  const float* __restrict__ ln_g,
                                  const float* __restrict__ ln_b,
                                  float* __restrict__ x3) {
  int t = blockIdx.x * blockDim.x + threadIdx.x;
  if (t >= NTOK) return;
  int b  = t / 3136, lt = t - b * 3136;
  int r  = lt / 56,  c  = lt - r * 56;
  int r0 = (r + 53) % 56, c0 = (c + 53) % 56;
  int q  = (r0 / 7) * 8 + (c0 / 7);
  int sl = q * 49 + (r0 % 7) * 7 + (c0 % 7);
  const float* src = x2 + ((size_t)b * 3136 + sl) * 96;
  const float* cur = x2 + (size_t)t * 96;
  float s = 0.f, ss = 0.f;
  for (int k = 0; k < 96; ++k) { float v = src[k]; s += v; ss += v * v; }
  float mean = s * (1.f / 96.f);
  float inv  = rsqrtf(ss * (1.f / 96.f) - mean * mean + LN_EPS);
  float* o = x3 + (size_t)t * 96;
  for (int k = 0; k < 96; ++k)
    o[k] = (src[k] - mean) * inv * ln_g[k] + ln_b[k] + cur[k];
}

// ---------------------------------------------------------------------------
// Host launcher. Input order (setup_inputs dict, params flattened in order):
// 0:x 1:ln1_g 2:ln1_b 3:ln2_g 4:ln2_b 5:ln3_g 6:ln3_b 7:ln4_g 8:ln4_b
// 9:w_qkv_w 10:w_qkv_b 11:w_proj_w 12:w_proj_b 13-16:sw_* (unused)
// 17:mlp1_w1 18:mlp1_b1 19:mlp1_w2 20:mlp1_b2 21:mlp2_w1 22:mlp2_b1
// 23:mlp2_w2 24:mlp2_b2
// ---------------------------------------------------------------------------
extern "C" void kernel_launch(void* const* d_in, const int* in_sizes, int n_in,
                              void* d_out, int out_size, void* d_ws,
                              size_t ws_size, hipStream_t stream) {
  (void)in_sizes; (void)n_in; (void)out_size; (void)ws_size;
  const float* x      = (const float*)d_in[0];
  const float* ln1_g  = (const float*)d_in[1];
  const float* ln1_b  = (const float*)d_in[2];
  const float* ln2_g  = (const float*)d_in[3];
  const float* ln2_b  = (const float*)d_in[4];
  const float* ln3_g  = (const float*)d_in[5];
  const float* ln3_b  = (const float*)d_in[6];
  const float* ln4_g  = (const float*)d_in[7];
  const float* ln4_b  = (const float*)d_in[8];
  const float* qkv_w  = (const float*)d_in[9];
  const float* qkv_b  = (const float*)d_in[10];
  const float* proj_w = (const float*)d_in[11];
  const float* proj_b = (const float*)d_in[12];
  const float* m1w1   = (const float*)d_in[17];
  const float* m1b1   = (const float*)d_in[18];
  const float* m1w2   = (const float*)d_in[19];
  const float* m1b2   = (const float*)d_in[20];
  const float* m2w1   = (const float*)d_in[21];
  const float* m2b1   = (const float*)d_in[22];
  const float* m2w2   = (const float*)d_in[23];
  const float* m2b2   = (const float*)d_in[24];

  char* ws = (char*)d_ws;
  float* x1 = (float*)(ws);
  float* x2 = (float*)(ws + XBYTES);
  float* x3 = (float*)(ws + 2 * XBYTES);
  _Float16* qkv_wh  = (_Float16*)(ws + 3 * XBYTES);
  _Float16* proj_wh = qkv_wh + 288 * 96;
  _Float16* m1w1h   = proj_wh + 96 * 96;
  _Float16* m1w2h   = m1w1h + 384 * 96;
  _Float16* m2w1h   = m1w2h + 96 * 384;
  _Float16* m2w2h   = m2w1h + 384 * 96;

  // weight conversion (tiny)
  cvt_f16_kernel<<<(288 * 96 + 255) / 256, 256, 0, stream>>>(qkv_w, qkv_wh, 288 * 96);
  cvt_f16_kernel<<<(96 * 96 + 255) / 256, 256, 0, stream>>>(proj_w, proj_wh, 96 * 96);
  cvt_f16_kernel<<<(384 * 96 + 255) / 256, 256, 0, stream>>>(m1w1, m1w1h, 384 * 96);
  cvt_f16_kernel<<<(96 * 384 + 255) / 256, 256, 0, stream>>>(m1w2, m1w2h, 96 * 384);
  cvt_f16_kernel<<<(384 * 96 + 255) / 256, 256, 0, stream>>>(m2w1, m2w1h, 384 * 96);
  cvt_f16_kernel<<<(96 * 384 + 255) / 256, 256, 0, stream>>>(m2w2, m2w2h, 96 * 384);

  // W-MSA block: x -> x1
  win_attn_kernel<<<4096, 128, 0, stream>>>(x, ln1_g, ln1_b, qkv_wh, qkv_b,
                                            proj_wh, proj_b, x1);
  // MLP1 (residual = original x): x1 -> x2
  mlp_kernel<<<NTOK / 64, 128, 0, stream>>>(x1, x, ln2_g, ln2_b, m1w1h, m1b1,
                                            m1w2h, m1b2, x2);
  // permutation + LN3 + residual: x2 -> x3
  permute_ln_kernel<<<(NTOK + 255) / 256, 256, 0, stream>>>(x2, ln3_g, ln3_b, x3);
  // MLP2 (residual = x3): x3 -> out
  mlp_kernel<<<NTOK / 64, 128, 0, stream>>>(x3, x3, ln4_g, ln4_b, m2w1h, m2b1,
                                            m2w2h, m2b2, (float*)d_out);
}
